// MultiHeadAttention_51049981280964
// MI455X (gfx1250) — matrix operand order repaired and checked by reading, not compiled, not run on hardware
//
#include <hip/hip_runtime.h>
#include <hip/hip_bf16.h>
#include <cstdint>

// ---------------------------------------------------------------------------
// MI455X (gfx1250) multi-head attention forward.
//   B=2, S=2048, D=1024, H=16, Dh=64.  All GEMMs via v_wmma_f32_16x16x32_bf16.
//   bf16 tile staging uses global_load_async_to_lds_b128 (ASYNCcnt-tracked).
// ---------------------------------------------------------------------------

typedef __attribute__((ext_vector_type(16))) __bf16        v16bf;
typedef __attribute__((ext_vector_type(8)))  float         v8f;
typedef __attribute__((ext_vector_type(4)))  float         f32x4;
typedef __attribute__((ext_vector_type(4)))  unsigned int  u32x4;
typedef __attribute__((ext_vector_type(4)))  int           i32x4;

// Address-space-qualified i32x4 for the async-to-LDS builtin
typedef __attribute__((address_space(1))) i32x4 as1_i32x4;
typedef __attribute__((address_space(3))) i32x4 as3_i32x4;

union Frag16 {
    v16bf v;
    u32x4 q[2];
};
union Pack16 {
    __bf16 h[16];
    u32x4  q[2];
};

// ---- gfx1250 async-to-LDS support (guarded; falls back to vector copies) ----
#if defined(__has_builtin)
#  if __has_builtin(__builtin_amdgcn_global_load_async_to_lds_b128)
#    define HAVE_ASYNC_B128 1
#  endif
#  if __has_builtin(__builtin_amdgcn_s_wait_asynccnt)
#    define WAIT_ASYNC() __builtin_amdgcn_s_wait_asynccnt(0)
#  endif
#endif
#ifndef HAVE_ASYNC_B128
#  define HAVE_ASYNC_B128 0
#endif
#ifndef WAIT_ASYNC
#  define WAIT_ASYNC() asm volatile("s_wait_asynccnt 0x0" ::: "memory")
#endif

__device__ __forceinline__ void copy16_to_lds(const __bf16* g, __bf16* l) {
#if HAVE_ASYNC_B128
    // AS1 pointer: 64-bit value reinterpreted; AS3 pointer: LDS aperture maps
    // addr[31:0] to the LDS byte address (ISA 10.2), so truncate.
    as1_i32x4* gp = (as1_i32x4*)(unsigned long long)(uintptr_t)g;
    as3_i32x4* lp = (as3_i32x4*)(unsigned int)(uintptr_t)l;
    __builtin_amdgcn_global_load_async_to_lds_b128(gp, lp, 0, 0);
#else
    *(u32x4*)l = *(const u32x4*)g;
#endif
}

__device__ __forceinline__ v8f vzero8() {
    v8f z = {};
    return z;
}

// A-fragment (16x32 bf16): lane = row (l16), halves = K = hi16*16 .. +15
// (contiguous, matching load_fragB).
__device__ __forceinline__ void load_fragA(Frag16& fr, const __bf16* rowp, int hi16) {
    fr.q[0] = *(const u32x4*)(rowp + hi16 * 16);
    fr.q[1] = *(const u32x4*)(rowp + hi16 * 16 + 8);
}
// B-fragment (32x16 bf16): lane = col (l16), halves = K = hi16*16 .. +15.
__device__ __forceinline__ void load_fragB(Frag16& fr, const __bf16* colp_hi) {
    fr.q[0] = *(const u32x4*)(colp_hi);
    fr.q[1] = *(const u32x4*)(colp_hi + 8);
}

// ---------------------------------------------------------------------------
// GEMM:  out[m,n] = scale * sum_k A[m,k] * W[n,k];  M=4096, N=K=1024.
// Block tile 128x128, 8 waves each owning 32x64 (2x4 WMMA tiles).
//   A_IS_F32: A fp32 (x), output bf16 in [B,H,S,Dh] head layout (Q/K/V).
//  !A_IS_F32: A bf16 (ctx, async-staged), output fp32 [M,N] (final proj).
// ---------------------------------------------------------------------------
template <bool A_IS_F32>
__global__ __launch_bounds__(256) void gemm_wmma_kernel(
    const float* __restrict__ A32,
    const __bf16* __restrict__ Abf,
    const float* __restrict__ W,
    __bf16* __restrict__ out_bf,
    float* __restrict__ out_f32,
    float out_scale)
{
    __shared__ __bf16 Ash[128 * 32];
    __shared__ __bf16 Bsh[128 * 32];

    const int tid   = threadIdx.x;
    const int wid   = tid >> 5;
    const int lane  = tid & 31;
    const int hi16  = lane >> 4;
    const int l16   = lane & 15;
    const int waveM = wid & 3;
    const int waveN = wid >> 2;
    const int row0  = blockIdx.x * 128;
    const int col0  = blockIdx.y * 128;
    const int sr    = tid >> 1;          // staging row 0..127
    const int sc    = (tid & 1) << 4;    // staging col 0 or 16

    v8f acc[2][4];
#pragma unroll
    for (int mt = 0; mt < 2; ++mt)
#pragma unroll
        for (int nt = 0; nt < 4; ++nt) acc[mt][nt] = vzero8();

    for (int kb = 0; kb < 1024; kb += 32) {
        __syncthreads();
        // ---- stage A tile (128x32 bf16) ----
        if constexpr (A_IS_F32) {
            const float* ap = &A32[(size_t)(row0 + sr) * 1024 + kb + sc];
            f32x4 f0 = *(const f32x4*)(ap + 0);
            f32x4 f1 = *(const f32x4*)(ap + 4);
            f32x4 f2 = *(const f32x4*)(ap + 8);
            f32x4 f3 = *(const f32x4*)(ap + 12);
            Pack16 pk;
#pragma unroll
            for (int i = 0; i < 4; ++i) {
                pk.h[i]      = (__bf16)f0[i];
                pk.h[4 + i]  = (__bf16)f1[i];
                pk.h[8 + i]  = (__bf16)f2[i];
                pk.h[12 + i] = (__bf16)f3[i];
            }
            *(u32x4*)&Ash[sr * 32 + sc]     = pk.q[0];
            *(u32x4*)&Ash[sr * 32 + sc + 8] = pk.q[1];
        } else {
            // bf16 source: async DMA to LDS (512 x 16B chunks, 2 per thread)
#pragma unroll
            for (int it = 0; it < 2; ++it) {
                int idx = tid + it * 256;
                int r = idx >> 2, p = (idx & 3) * 8;
                copy16_to_lds(&Abf[(size_t)(row0 + r) * 1024 + kb + p],
                              &Ash[r * 32 + p]);
            }
        }
        // ---- stage B = W tile (128x32 fp32 -> bf16) ----
        {
            const float* wp = &W[(size_t)(col0 + sr) * 1024 + kb + sc];
            f32x4 g0 = *(const f32x4*)(wp + 0);
            f32x4 g1 = *(const f32x4*)(wp + 4);
            f32x4 g2 = *(const f32x4*)(wp + 8);
            f32x4 g3 = *(const f32x4*)(wp + 12);
            Pack16 pk;
#pragma unroll
            for (int i = 0; i < 4; ++i) {
                pk.h[i]      = (__bf16)g0[i];
                pk.h[4 + i]  = (__bf16)g1[i];
                pk.h[8 + i]  = (__bf16)g2[i];
                pk.h[12 + i] = (__bf16)g3[i];
            }
            *(u32x4*)&Bsh[sr * 32 + sc]     = pk.q[0];
            *(u32x4*)&Bsh[sr * 32 + sc + 8] = pk.q[1];
        }
        // prefetch next K slab into GL2
        if (kb + 32 < 1024) {
            if constexpr (A_IS_F32)
                __builtin_prefetch(&A32[(size_t)(row0 + sr) * 1024 + kb + 32 + sc], 0, 0);
            __builtin_prefetch(&W[(size_t)(col0 + sr) * 1024 + kb + 32 + sc], 0, 0);
        }
#if HAVE_ASYNC_B128
        if constexpr (!A_IS_F32) WAIT_ASYNC();
#endif
        __syncthreads();

        // ---- fragments + WMMA ----
        Frag16 afr[2], bfr[4];
#pragma unroll
        for (int mt = 0; mt < 2; ++mt)
            load_fragA(afr[mt], &Ash[(waveM * 32 + mt * 16 + l16) * 32], hi16);
#pragma unroll
        for (int nt = 0; nt < 4; ++nt)
            load_fragB(bfr[nt], &Bsh[(waveN * 64 + nt * 16 + l16) * 32 + hi16 * 16]);
#pragma unroll
        for (int mt = 0; mt < 2; ++mt)
#pragma unroll
            for (int nt = 0; nt < 4; ++nt)
                acc[mt][nt] = __builtin_amdgcn_wmma_f32_16x16x32_bf16(
                    false, afr[mt].v, false, bfr[nt].v,
                    (short)0, acc[mt][nt], false, false);
    }

    // ---- epilogue (C/D layout: m = 8*hi16 + i, n = l16) ----
#pragma unroll
    for (int mt = 0; mt < 2; ++mt)
#pragma unroll
        for (int nt = 0; nt < 4; ++nt)
#pragma unroll
            for (int i = 0; i < 8; ++i) {
                int m = row0 + waveM * 32 + mt * 16 + hi16 * 8 + i;
                int n = col0 + waveN * 64 + nt * 16 + l16;
                float vv = acc[mt][nt][i] * out_scale;
                if constexpr (A_IS_F32) {
                    int bb = m >> 11, s = m & 2047, hh = n >> 6, d = n & 63;
                    out_bf[((((size_t)bb * 16 + hh) * 2048 + s) * 64) + d] = (__bf16)vv;
                } else {
                    out_f32[(size_t)m * 1024 + n] = vv;
                }
            }
}

// ---------------------------------------------------------------------------
// Flash attention: grid (S/128, H, B), 8 waves; each wave owns 16 q-rows with
// the full 16x64 fp32 accumulator in registers.  K async-staged to LDS,
// V transposed at staging, online softmax via wave32 shfl_xor butterflies,
// P re-laid out C->A through a per-wave LDS patch.
// ---------------------------------------------------------------------------
__global__ __launch_bounds__(256) void flash_attn_kernel(
    const __bf16* __restrict__ Q, const __bf16* __restrict__ K,
    const __bf16* __restrict__ V, __bf16* __restrict__ ctx)
{
    __shared__ __bf16 Ksh[64 * 64];        // [s_local][d]
    __shared__ __bf16 Vsh[64 * 64];        // [d][s_local]
    __shared__ __bf16 Psh[8][16 * 64];     // per-wave P patch

    const int tid  = threadIdx.x;
    const int wid  = tid >> 5;
    const int lane = tid & 31;
    const int hi16 = lane >> 4;
    const int l16  = lane & 15;
    const int b    = blockIdx.z;
    const int h    = blockIdx.y;
    const int q0   = blockIdx.x * 128 + wid * 16;

    const size_t head_off = ((size_t)b * 16 + h) * 2048 * 64;
    const __bf16* Qb = Q + head_off;
    const __bf16* Kb = K + head_off;
    const __bf16* Vb = V + head_off;

    Frag16 qa[2];
#pragma unroll
    for (int c = 0; c < 2; ++c)
        load_fragA(qa[c], Qb + (size_t)(q0 + l16) * 64 + c * 32, hi16);

    v8f O[4];
#pragma unroll
    for (int nt = 0; nt < 4; ++nt) O[nt] = vzero8();
    float mrow[8], lrow[8];
#pragma unroll
    for (int i = 0; i < 8; ++i) { mrow[i] = -3.0e38f; lrow[i] = 0.0f; }

    for (int s0 = 0; s0 < 2048; s0 += 64) {
        __syncthreads();
        // ---- stage K tile: 64x64 bf16 = 512 x 16B, async when available ----
#pragma unroll
        for (int it = 0; it < 2; ++it) {
            int idx = (tid + it * 256) * 8;   // bf16 offset, 16B-aligned
            copy16_to_lds(Kb + (size_t)s0 * 64 + idx, &Ksh[idx]);
        }
        // ---- stage V tile transposed: Vsh[d][s] ----
#pragma unroll
        for (int it = 0; it < 2; ++it) {
            int idx = tid + it * 256;          // 512 x 16B chunks
            int s = idx >> 3, grp = (idx & 7) * 8;
            union { u32x4 q; __bf16 hh[8]; } cv;
            cv.q = *(const u32x4*)(Vb + (size_t)(s0 + s) * 64 + grp);
#pragma unroll
            for (int e = 0; e < 8; ++e)
                Vsh[(grp + e) * 64 + s] = cv.hh[e];
        }
#if HAVE_ASYNC_B128
        WAIT_ASYNC();
#endif
        __syncthreads();

        // ---- scores: S[16x64] = Q * K^T ----
        v8f S4[4];
#pragma unroll
        for (int nt = 0; nt < 4; ++nt) S4[nt] = vzero8();
#pragma unroll
        for (int nt = 0; nt < 4; ++nt)
#pragma unroll
            for (int c = 0; c < 2; ++c) {
                Frag16 bk;
                load_fragB(bk, &Ksh[(nt * 16 + l16) * 64 + c * 32 + hi16 * 16]);
                S4[nt] = __builtin_amdgcn_wmma_f32_16x16x32_bf16(
                    false, qa[c].v, false, bk.v, (short)0, S4[nt], false, false);
            }

        // ---- online softmax (row = hi16*8 + i, col group = l16) ----
#pragma unroll
        for (int i = 0; i < 8; ++i) {
            float rm = fmaxf(fmaxf(S4[0][i], S4[1][i]), fmaxf(S4[2][i], S4[3][i]));
            rm = fmaxf(rm, __shfl_xor(rm, 1, 32));
            rm = fmaxf(rm, __shfl_xor(rm, 2, 32));
            rm = fmaxf(rm, __shfl_xor(rm, 4, 32));
            rm = fmaxf(rm, __shfl_xor(rm, 8, 32));
            float mnew = fmaxf(mrow[i], rm);
            float corr = __expf(mrow[i] - mnew);
            float rs = 0.0f;
#pragma unroll
            for (int nt = 0; nt < 4; ++nt) {
                float p = __expf(S4[nt][i] - mnew);
                S4[nt][i] = p;
                rs += p;
            }
            rs += __shfl_xor(rs, 1, 32);
            rs += __shfl_xor(rs, 2, 32);
            rs += __shfl_xor(rs, 4, 32);
            rs += __shfl_xor(rs, 8, 32);
            lrow[i] = lrow[i] * corr + rs;
            mrow[i] = mnew;
#pragma unroll
            for (int nt = 0; nt < 4; ++nt) O[nt][i] *= corr;
        }

        // ---- P: C/D layout -> A layout via per-wave LDS patch ----
#pragma unroll
        for (int nt = 0; nt < 4; ++nt)
#pragma unroll
            for (int i = 0; i < 8; ++i)
                Psh[wid][(hi16 * 8 + i) * 64 + nt * 16 + l16] = (__bf16)S4[nt][i];
        // same-wave LDS ops are in-order; no workgroup barrier needed

        // ---- O += P (16x64) * V (64x64) ----
#pragma unroll
        for (int c = 0; c < 2; ++c) {
            Frag16 pa;
            load_fragA(pa, &Psh[wid][(size_t)l16 * 64 + c * 32], hi16);
#pragma unroll
            for (int nt = 0; nt < 4; ++nt) {
                Frag16 bv;
                load_fragB(bv, &Vsh[(nt * 16 + l16) * 64 + c * 32 + hi16 * 16]);
                O[nt] = __builtin_amdgcn_wmma_f32_16x16x32_bf16(
                    false, pa.v, false, bv.v, (short)0, O[nt], false, false);
            }
        }
    }

    // ---- epilogue: ctx[b, q, h*64 + d] = O / l ----
#pragma unroll
    for (int nt = 0; nt < 4; ++nt)
#pragma unroll
        for (int i = 0; i < 8; ++i) {
            int q = q0 + hi16 * 8 + i;
            int d = nt * 16 + l16;
            ctx[((size_t)b * 2048 + q) * 1024 + h * 64 + d] =
                (__bf16)(O[nt][i] / lrow[i]);
        }
}

// ---------------------------------------------------------------------------
extern "C" void kernel_launch(void* const* d_in, const int* in_sizes, int n_in,
                              void* d_out, int out_size, void* d_ws, size_t ws_size,
                              hipStream_t stream) {
    (void)in_sizes; (void)n_in; (void)out_size; (void)ws_size;
    const float* x  = (const float*)d_in[0];
    const float* Wq = (const float*)d_in[1];
    const float* Wk = (const float*)d_in[2];
    const float* Wv = (const float*)d_in[3];
    const float* Wo = (const float*)d_in[4];
    float* out = (float*)d_out;

    const size_t ELEMS = (size_t)4 * 1024 * 1024;  // 4096 x 1024
    __bf16* Qw = (__bf16*)d_ws;
    __bf16* Kw = Qw + ELEMS;
    __bf16* Vw = Kw + ELEMS;
    __bf16* Cw = Vw + ELEMS;

    const float SCALE = 0.125f;  // 1/sqrt(64)

    dim3 blk(256);
    dim3 gemm_grid(32, 8);
    gemm_wmma_kernel<true><<<gemm_grid, blk, 0, stream>>>(x, nullptr, Wq, Qw, nullptr, SCALE);
    gemm_wmma_kernel<true><<<gemm_grid, blk, 0, stream>>>(x, nullptr, Wk, Kw, nullptr, 1.0f);
    gemm_wmma_kernel<true><<<gemm_grid, blk, 0, stream>>>(x, nullptr, Wv, Vw, nullptr, 1.0f);

    flash_attn_kernel<<<dim3(16, 16, 2), blk, 0, stream>>>(Qw, Kw, Vw, Cw);

    gemm_wmma_kernel<false><<<gemm_grid, blk, 0, stream>>>(nullptr, Cw, Wo, nullptr, out, 1.0f);
}
